// QuantumKernelMethod_65481071405937
// MI455X (gfx1250) — matrix operand order repaired and checked by reading, not compiled, not run on hardware
//
#include <hip/hip_runtime.h>
#include <hip/hip_bf16.h>

typedef __attribute__((ext_vector_type(16))) _Float16 v16h;
typedef __attribute__((ext_vector_type(8)))  _Float16 v8h;
typedef __attribute__((ext_vector_type(8)))  float    v8f;

#define WMMA_F16(A, B, C) \
    __builtin_amdgcn_wmma_f32_16x16x32_f16(false, (A), false, (B), (short)0, (C), false, false)

#define LDS_LD 40   // P-tile leading dim in halves: 80B = 20 banks, conflict-free & 16B aligned

// ---------------------------------------------------------------------------
// A-operand fragment (16x32 f16), CDNA5 A layout:
//   lanes 0-15 : row m = lane,    K = {0..7} U {16..23}
//   lanes 16-31: row m = lane-16, K = {8..15} U {24..31}
__device__ __forceinline__ v16h load_a_frag(const _Float16* base, int ldh, int lane) {
    int m = lane & 15;
    int o = (lane & 16) ? 8 : 0;
    const _Float16* r = base + m * ldh + o;
    union { v16h v; v8h h[2]; } u;
    u.h[0] = *(const v8h*)(r);
    u.h[1] = *(const v8h*)(r + 16);
    return u.v;
}

// B-operand fragment (32x16 f16), loaded from B^T rows (row-major, ld `ldh`):
//   lane n (0-15): K = {0..7} U {16..23} of column n; lanes 16-31: K = {8..15} U {24..31}
__device__ __forceinline__ v16h load_b_frag(const _Float16* base, int ldh, int lane) {
    int n = lane & 15;
    int o = (lane & 16) ? 8 : 0;
    const _Float16* r = base + n * ldh + o;
    union { v16h v; v8h h[2]; } u;
    u.h[0] = *(const v8h*)(r);
    u.h[1] = *(const v8h*)(r + 16);
    return u.v;
}

// ---------------------------------------------------------------------------
// prep: Qh = (X @ Wq) * (1/sqrt(64)) f16, Kh = X @ Wk f16, XT = X^T f16.
__global__ __launch_bounds__(256) void prep_kernel(
    const float* __restrict__ X, const float* __restrict__ Wq,
    const float* __restrict__ Wk, _Float16* __restrict__ Qh,
    _Float16* __restrict__ Kh, _Float16* __restrict__ XT, int N) {
    __shared__ float xs[4][64];
    int lr  = threadIdx.x >> 6;
    int c   = threadIdx.x & 63;
    int row = blockIdx.x * 4 + lr;
    float xc = X[row * 64 + c];
    xs[lr][c] = xc;
    __syncthreads();
    const float* xrow = xs[lr];
    float q = 0.f, k = 0.f;
#pragma unroll
    for (int t = 0; t < 64; ++t) {
        float xv = xrow[t];
        q = fmaf(xv, Wq[t * 64 + c], q);
        k = fmaf(xv, Wk[t * 64 + c], k);
    }
    Qh[row * 64 + c] = (_Float16)(q * 0.125f);   // fold 1/sqrt(d)
    Kh[row * 64 + c] = (_Float16)k;
    XT[c * N + row]  = (_Float16)xc;
}

// ---------------------------------------------------------------------------
// Flash attention, transposed-score formulation. One wave = 16 query rows.
// Scores computed as S^T = K_tile @ Q^T so C-fragment rows (VGPR dim) = keys,
// lanes = queries -> softmax reductions are in-lane + one xor-16 shuffle.
// Handles both x and y in one grid (blocks [0,nblkx) -> x, rest -> y).
__global__ __launch_bounds__(128) void attn_kernel(
    const _Float16* __restrict__ Qhx, const _Float16* __restrict__ Khx,
    const _Float16* __restrict__ XTx, _Float16* __restrict__ ATTx,
    float* __restrict__ NRMx, int Nx,
    const _Float16* __restrict__ Qhy, const _Float16* __restrict__ Khy,
    const _Float16* __restrict__ XTy, _Float16* __restrict__ ATTy,
    float* __restrict__ NRMy, int Ny, int nblkx) {
    __shared__ __align__(16) _Float16 Pld[4][16 * LDS_LD];
    int lane = threadIdx.x & 31;
    int wave = threadIdx.x >> 5;
    int hi   = lane >> 4;
    int n    = lane & 15;              // query column within tile

    int bid = blockIdx.x;
    const _Float16 *Qh, *Kh, *XT;
    _Float16* ATT;
    float* NRM;
    int N;
    if (bid < nblkx) { Qh = Qhx; Kh = Khx; XT = XTx; ATT = ATTx; NRM = NRMx; N = Nx; }
    else { bid -= nblkx; Qh = Qhy; Kh = Khy; XT = XTy; ATT = ATTy; NRM = NRMy; N = Ny; }

    int qrow0 = (bid * 4 + wave) * 16;
    _Float16* myP = Pld[wave];

    // Q^T as B operands (loop-invariant)
    v16h bQ0 = load_b_frag(Qh + qrow0 * 64, 64, lane);       // K 0..31
    v16h bQ1 = load_b_frag(Qh + qrow0 * 64 + 32, 64, lane);  // K 32..63

    v8f acc0 = {}, acc1 = {}, acc2 = {}, acc3 = {};
    float mrun = -1e30f, lrun = 0.f;

    for (int kb = 0; kb < N; kb += 32) {
        if (kb + 32 < N) {
            // one line per key row (row = 128B); 32 lanes cover the whole next block
            __builtin_prefetch(Kh + (size_t)(kb + 32 + lane) * 64, 0, 0);
            __builtin_prefetch(XT + (size_t)(2 * lane) * N + kb + 32, 0, 0);
            __builtin_prefetch(XT + (size_t)(2 * lane + 1) * N + kb + 32, 0, 0);
        }
        // issue all global loads for this iteration up front
        v16h aK00 = load_a_frag(Kh + (size_t)kb * 64, 64, lane);            // keys kb..+15, k 0..31
        v16h aK01 = load_a_frag(Kh + (size_t)kb * 64 + 32, 64, lane);       // keys kb..+15, k 32..63
        v16h aK10 = load_a_frag(Kh + (size_t)(kb + 16) * 64, 64, lane);     // keys kb+16..+31
        v16h aK11 = load_a_frag(Kh + (size_t)(kb + 16) * 64 + 32, 64, lane);
        v16h bV0  = load_b_frag(XT + 0  * (size_t)N + kb, N, lane);
        v16h bV1  = load_b_frag(XT + 16 * (size_t)N + kb, N, lane);
        v16h bV2  = load_b_frag(XT + 32 * (size_t)N + kb, N, lane);
        v16h bV3  = load_b_frag(XT + 48 * (size_t)N + kb, N, lane);

        // S^T tiles: ST*[p] = score(key, query=n), key = p + 8*hi (+16 for ST1)
        v8f z = {};
        v8f ST0 = WMMA_F16(aK00, bQ0, z);
        ST0     = WMMA_F16(aK01, bQ1, ST0);
        v8f ST1 = WMMA_F16(aK10, bQ0, z);
        ST1     = WMMA_F16(aK11, bQ1, ST1);

        // per-query (per-lane) online softmax
        float t = fmaxf(ST0[0], ST1[0]);
#pragma unroll
        for (int p = 1; p < 8; ++p) t = fmaxf(t, fmaxf(ST0[p], ST1[p]));
        t = fmaxf(t, __shfl_xor(t, 16, 32));
        float mnew = fmaxf(mrun, t);

        float e0[8], e1[8];
        float rs = 0.f;
#pragma unroll
        for (int p = 0; p < 8; ++p) {
            e0[p] = __expf(ST0[p] - mnew);
            e1[p] = __expf(ST1[p] - mnew);
            rs += e0[p] + e1[p];
        }
        rs += __shfl_xor(rs, 16, 32);
        float corr = __expf(mrun - mnew);
        lrun = lrun * corr + rs;
        mrun = mnew;

        // P tile to LDS, row-major [query][key]: keys contiguous -> 2x b128 store
        v8h pk0, pk1;
#pragma unroll
        for (int p = 0; p < 8; ++p) { pk0[p] = (_Float16)e0[p]; pk1[p] = (_Float16)e1[p]; }
        *(v8h*)(myP + n * LDS_LD + 8 * hi)      = pk0;   // keys 8*hi .. 8*hi+7
        *(v8h*)(myP + n * LDS_LD + 16 + 8 * hi) = pk1;   // keys 16+8*hi ..

        // rescale accumulator rows (row = query p + 8*hi) by that query's corr
#pragma unroll
        for (int p = 0; p < 8; ++p) {
            float c = __shfl(corr, p + 8 * hi, 32);
            acc0[p] *= c; acc1[p] *= c; acc2[p] *= c; acc3[p] *= c;
        }

        v16h aP = load_a_frag(myP, LDS_LD, lane);
        acc0 = WMMA_F16(aP, bV0, acc0);
        acc1 = WMMA_F16(aP, bV1, acc1);
        acc2 = WMMA_F16(aP, bV2, acc2);
        acc3 = WMMA_F16(aP, bV3, acc3);
    }

    // epilogue: normalize, write f16 attention + f32 squared norms
    float inv = 1.0f / lrun;   // per query (lane)
#pragma unroll
    for (int p = 0; p < 8; ++p) {
        float iv = __shfl(inv, p + 8 * hi, 32);   // inv for query row p+8*hi
        int row = qrow0 + p + hi * 8;
        float o0 = acc0[p] * iv, o1 = acc1[p] * iv;
        float o2 = acc2[p] * iv, o3 = acc3[p] * iv;
        ATT[row * 64 + n]      = (_Float16)o0;
        ATT[row * 64 + 16 + n] = (_Float16)o1;
        ATT[row * 64 + 32 + n] = (_Float16)o2;
        ATT[row * 64 + 48 + n] = (_Float16)o3;
        float sq = o0 * o0 + o1 * o1 + o2 * o2 + o3 * o3;
        sq += __shfl_xor(sq, 1, 32);
        sq += __shfl_xor(sq, 2, 32);
        sq += __shfl_xor(sq, 4, 32);
        sq += __shfl_xor(sq, 8, 32);
        if (n == 0) NRM[row] = sq;
    }
}

// ---------------------------------------------------------------------------
// rbf: out[i][j] = exp(2*(ax_i . ay_j) - |ax_i|^2 - |ay_j|^2), NT streaming store.
__global__ __launch_bounds__(256) void rbf_kernel(
    const _Float16* __restrict__ AX, const float* __restrict__ NXn,
    const _Float16* __restrict__ AY, const float* __restrict__ NYn,
    float* __restrict__ out, int Ny) {
    int lane = threadIdx.x & 31;
    int wave = threadIdx.x >> 5;
    int hi   = lane >> 4;
    int n    = lane & 15;
    int row0 = blockIdx.y * 128 + wave * 16;
    int col0 = blockIdx.x * 64;

    v16h a0 = load_a_frag(AX + row0 * 64, 64, lane);       // K 0..31
    v16h a1 = load_a_frag(AX + row0 * 64 + 32, 64, lane);  // K 32..63

    v8f c0 = {}, c1 = {}, c2 = {}, c3 = {};
    v16h b;
    b = load_b_frag(AY + (col0 +  0) * 64,      64, lane); c0 = WMMA_F16(a0, b, c0);
    b = load_b_frag(AY + (col0 +  0) * 64 + 32, 64, lane); c0 = WMMA_F16(a1, b, c0);
    b = load_b_frag(AY + (col0 + 16) * 64,      64, lane); c1 = WMMA_F16(a0, b, c1);
    b = load_b_frag(AY + (col0 + 16) * 64 + 32, 64, lane); c1 = WMMA_F16(a1, b, c1);
    b = load_b_frag(AY + (col0 + 32) * 64,      64, lane); c2 = WMMA_F16(a0, b, c2);
    b = load_b_frag(AY + (col0 + 32) * 64 + 32, 64, lane); c2 = WMMA_F16(a1, b, c2);
    b = load_b_frag(AY + (col0 + 48) * 64,      64, lane); c3 = WMMA_F16(a0, b, c3);
    b = load_b_frag(AY + (col0 + 48) * 64 + 32, 64, lane); c3 = WMMA_F16(a1, b, c3);

    float ny0 = NYn[col0 + n];
    float ny1 = NYn[col0 + 16 + n];
    float ny2 = NYn[col0 + 32 + n];
    float ny3 = NYn[col0 + 48 + n];
#pragma unroll
    for (int p = 0; p < 8; ++p) {
        int i = row0 + p + hi * 8;
        float nx = NXn[i];
        float* o = out + (size_t)i * Ny + col0 + n;   // coalesced across lanes
        __builtin_nontemporal_store(__expf(2.f * c0[p] - nx - ny0), o + 0);
        __builtin_nontemporal_store(__expf(2.f * c1[p] - nx - ny1), o + 16);
        __builtin_nontemporal_store(__expf(2.f * c2[p] - nx - ny2), o + 32);
        __builtin_nontemporal_store(__expf(2.f * c3[p] - nx - ny3), o + 48);
    }
}

// ---------------------------------------------------------------------------
extern "C" void kernel_launch(void* const* d_in, const int* in_sizes, int n_in,
                              void* d_out, int out_size, void* d_ws, size_t ws_size,
                              hipStream_t stream) {
    const float* Wq = (const float*)d_in[0];  // rotation_params
    const float* Wk = (const float*)d_in[1];  // entangle_params
    const float* x  = (const float*)d_in[2];
    const float* y  = (const float*)d_in[3];
    float* out = (float*)d_out;
    const int D  = 64;
    const int Nx = in_sizes[2] / D;
    const int Ny = in_sizes[3] / D;

    char* ws = (char*)d_ws;
    size_t off = 0;
    auto carve = [&](size_t bytes) -> char* {
        char* p = ws + off;
        off += (bytes + 255) & ~(size_t)255;
        return p;
    };
    _Float16* Qx = (_Float16*)carve((size_t)Nx * D * 2);
    _Float16* Kx = (_Float16*)carve((size_t)Nx * D * 2);
    _Float16* XT = (_Float16*)carve((size_t)Nx * D * 2);
    _Float16* AX = (_Float16*)carve((size_t)Nx * D * 2);
    float*   NXn = (float*)carve((size_t)Nx * 4);
    _Float16* Qy = (_Float16*)carve((size_t)Ny * D * 2);
    _Float16* Ky = (_Float16*)carve((size_t)Ny * D * 2);
    _Float16* YT = (_Float16*)carve((size_t)Ny * D * 2);
    _Float16* AY = (_Float16*)carve((size_t)Ny * D * 2);
    float*   NYn = (float*)carve((size_t)Ny * 4);

    prep_kernel<<<Nx / 4, 256, 0, stream>>>(x, Wq, Wk, Qx, Kx, XT, Nx);
    prep_kernel<<<Ny / 4, 256, 0, stream>>>(y, Wq, Wk, Qy, Ky, YT, Ny);
    int nblkx = Nx / 64;
    int nblky = Ny / 64;
    attn_kernel<<<nblkx + nblky, 128, 0, stream>>>(Qx, Kx, XT, AX, NXn, Nx,
                                                   Qy, Ky, YT, AY, NYn, Ny, nblkx);
    dim3 grid(Ny / 64, Nx / 128);
    rbf_kernel<<<grid, 256, 0, stream>>>(AX, NXn, AY, NYn, out, Ny);
}
